// SimpleMambaLayer_12068858101826
// MI455X (gfx1250) — matrix operand order repaired and checked by reading, not compiled, not run on hardware
//
#include <hip/hip_runtime.h>
#include <hip/hip_bf16.h>
#include <math.h>

// ---------------- problem constants ----------------
#define BATCH     8
#define DIM       256
#define L_SEQ     1024          // 32*32
#define D_INNER   512
#define D_STATE   16
#define DT_RANK   16
#define N_TOK     (BATCH * L_SEQ)   // 8192

typedef __attribute__((ext_vector_type(16))) _Float16 v16h;
typedef __attribute__((ext_vector_type(8)))  _Float16 v8h;
typedef __attribute__((ext_vector_type(8)))  float    v8f;

// ---------------------------------------------------------------------------
// WMMA 16x16 tile: A is row-major (M x K) f16, Bt is the TRANSPOSED weight,
// row-major (N x K) f16, so both fragment loads are contiguous 16B chunks.
// Fragment layouts per CDNA5 ISA 7.12.2 (wave32):
//   A v16h:  lane<16 -> K = kk+[0..7], kk+16+[0..7]; lane>=16 -> +8 on both
//   B v16h:  same K order as A: lane<16 -> K = kk+[0..7], kk+16+[0..7]; lane>=16 -> +8 on both
//   C/D v8f: c[r] = D[r + (lane>=16 ? 8 : 0)][lane&15]
// ---------------------------------------------------------------------------
__device__ __forceinline__ v8f wmma_tile_f16(const _Float16* __restrict__ A, int lda,
                                             const _Float16* __restrict__ Bt, int ldb,
                                             int mbase, int nbase, int K, int lane) {
    v8f c = {};
    const int aoff = (lane >> 4) << 3;   // 0 or 8 halfs
    const int boff = (lane >> 4) << 3;   // 0 or 8 halfs
    const _Float16* ap = A  + (size_t)(mbase + (lane & 15)) * lda + aoff;
    const _Float16* bp = Bt + (size_t)(nbase + (lane & 15)) * ldb + boff;
    for (int kk = 0; kk < K; kk += 32) {
        v8h a0 = *(const v8h*)(ap + kk);
        v8h a1 = *(const v8h*)(ap + kk + 16);
        v8h b0 = *(const v8h*)(bp + kk);
        v8h b1 = *(const v8h*)(bp + kk + 16);
        v16h a, b;
#pragma unroll
        for (int i = 0; i < 8; ++i) {
            a[i] = a0[i]; a[i + 8] = a1[i];
            b[i] = b0[i]; b[i + 8] = b1[i];
        }
        c = __builtin_amdgcn_wmma_f32_16x16x32_f16(
                /*neg_a=*/false, a, /*neg_b=*/false, b,
                /*c_mod=*/(short)0, c, /*reuse_a=*/false, /*reuse_b=*/false);
    }
    return c;
}

// ---------------------------------------------------------------------------
// K0: convert weights to f16 AND transpose to (N x K) so WMMA B loads are
// contiguous.  WhinT[n][k]=W_in[k][n], WhxpT[n][k]=W_xproj[k][n],
// WhoutT[n][k]=W_out[k][n].
// ---------------------------------------------------------------------------
#define E_WIN  (DIM * 2 * D_INNER)               // 256*1024
#define E_WXP  (D_INNER * (DT_RANK + 2*D_STATE)) // 512*48
#define E_WOUT (D_INNER * DIM)                   // 512*256

__global__ void convert_weights_kernel(const float* __restrict__ W_in,
                                       const float* __restrict__ W_xproj,
                                       const float* __restrict__ W_out,
                                       _Float16* __restrict__ WhinT,
                                       _Float16* __restrict__ WhxpT,
                                       _Float16* __restrict__ WhoutT) {
    int g = blockIdx.x * 256 + threadIdx.x;
    if (g < E_WIN) {
        int n = g / DIM, k = g % DIM;                 // dest [n][k], K=256
        WhinT[g] = (_Float16)W_in[(size_t)k * (2 * D_INNER) + n];
    } else if (g < E_WIN + E_WXP) {
        int g2 = g - E_WIN;
        int n = g2 / D_INNER, k = g2 % D_INNER;       // K=512
        WhxpT[g2] = (_Float16)W_xproj[(size_t)k * (DT_RANK + 2*D_STATE) + n];
    } else if (g < E_WIN + E_WXP + E_WOUT) {
        int g3 = g - E_WIN - E_WXP;
        int n = g3 / D_INNER, k = g3 % D_INNER;       // K=512
        WhoutT[g3] = (_Float16)W_out[(size_t)k * DIM + n];
    }
}

// ---------------------------------------------------------------------------
// K1: LayerNorm over C=256 per token, emit f16 activations (token-major).
// x is (B, C, H, W); token t=(b,l) reads x[b][c][l].
// ---------------------------------------------------------------------------
__global__ void layernorm_kernel(const float* __restrict__ x,
                                 const float* __restrict__ gamma,
                                 const float* __restrict__ beta,
                                 _Float16* __restrict__ xnh) {
    __shared__ float s1[DIM], s2[DIM];
    const int t = blockIdx.x;          // 0..8191
    const int c = threadIdx.x;         // 0..255
    const int b = t >> 10, l = t & (L_SEQ - 1);
    float v = x[(size_t)b * DIM * L_SEQ + (size_t)c * L_SEQ + l];
    s1[c] = v; s2[c] = v * v;
    __syncthreads();
    for (int s = 128; s > 0; s >>= 1) {
        if (c < s) { s1[c] += s1[c + s]; s2[c] += s2[c + s]; }
        __syncthreads();
    }
    float mu  = s1[0] * (1.0f / DIM);
    float var = s2[0] * (1.0f / DIM) - mu * mu;
    float xn = (v - mu) * rsqrtf(var + 1e-5f) * gamma[c] + beta[c];
    xnh[(size_t)t * DIM + c] = (_Float16)xn;
}

// ---------------------------------------------------------------------------
// K2: xz(8192x1024) = xnh(8192x256) @ W_in   (WMMA, K=256)
// One wave per 16x16 output tile, 8 waves/block.
// ---------------------------------------------------------------------------
__global__ void gemm_in_kernel(const _Float16* __restrict__ A,
                               const _Float16* __restrict__ Bt,
                               float* __restrict__ C) {
    const int lane = threadIdx.x & 31;
    const int tile = blockIdx.x * 8 + (threadIdx.x >> 5);
    const int NT = (2 * D_INNER) / 16;            // 64
    const int mt = tile / NT, nt = tile % NT;
    v8f c = wmma_tile_f16(A, DIM, Bt, DIM, mt * 16, nt * 16, DIM, lane);
    const int n = nt * 16 + (lane & 15);
    const int m0 = mt * 16 + ((lane >> 4) << 3);
#pragma unroll
    for (int r = 0; r < 8; ++r)
        C[(size_t)(m0 + r) * (2 * D_INNER) + n] = c[r];
}

// ---------------------------------------------------------------------------
// K3: depthwise causal conv (taps=4) + bias + SiLU along L, per (b,d).
// Reads u from xz cols [0,512); writes u2 (f32) and u2h (f16).
// ---------------------------------------------------------------------------
__global__ void conv_silu_kernel(const float* __restrict__ xz,
                                 const float* __restrict__ conv_w,
                                 const float* __restrict__ conv_b,
                                 float* __restrict__ u2,
                                 _Float16* __restrict__ u2h) {
    const int id = blockIdx.x * 256 + threadIdx.x;  // 0..4095
    const int b = id >> 9, d = id & (D_INNER - 1);
    const float w0 = conv_w[d * 4 + 0], w1 = conv_w[d * 4 + 1];
    const float w2 = conv_w[d * 4 + 2], w3 = conv_w[d * 4 + 3];
    const float bias = conv_b[d];
    const float* up = xz + (size_t)b * L_SEQ * (2 * D_INNER) + d;
    float*     o  = u2  + (size_t)b * L_SEQ * D_INNER + d;
    _Float16*  oh = u2h + (size_t)b * L_SEQ * D_INNER + d;
    float xm3 = 0.f, xm2 = 0.f, xm1 = 0.f;
    for (int l = 0; l < L_SEQ; ++l) {
        float cur = up[(size_t)l * (2 * D_INNER)];
        float a = w0 * xm3 + w1 * xm2 + w2 * xm1 + w3 * cur + bias;
        float s = a / (1.f + expf(-a));            // SiLU
        o[(size_t)l * D_INNER]  = s;
        oh[(size_t)l * D_INNER] = (_Float16)s;
        xm3 = xm2; xm2 = xm1; xm1 = cur;
    }
}

// ---------------------------------------------------------------------------
// K4: dbc(8192x48) = u2h(8192x512) @ W_xproj   (WMMA, K=512, 3 N-tiles)
// ---------------------------------------------------------------------------
__global__ void gemm_xproj_kernel(const _Float16* __restrict__ A,
                                  const _Float16* __restrict__ Bt,
                                  float* __restrict__ C) {
    const int lane = threadIdx.x & 31;
    const int tile = blockIdx.x * 8 + (threadIdx.x >> 5);
    const int NT = (DT_RANK + 2 * D_STATE) / 16;   // 3
    const int mt = tile / NT, nt = tile % NT;
    v8f c = wmma_tile_f16(A, D_INNER, Bt, D_INNER, mt * 16, nt * 16, D_INNER, lane);
    const int n = nt * 16 + (lane & 15);
    const int m0 = mt * 16 + ((lane >> 4) << 3);
#pragma unroll
    for (int r = 0; r < 8; ++r)
        C[(size_t)(m0 + r) * (DT_RANK + 2 * D_STATE) + n] = c[r];
}

// ---------------------------------------------------------------------------
// K5: selective scan (fused dt-projection + softplus + recurrence + D-skip +
// SiLU(z) gate).  One block per batch, one thread per channel d.
// dbc row (48 fp32) broadcast via LDS each step; W_dt column in registers.
// ---------------------------------------------------------------------------
__global__ void scan_kernel(const float* __restrict__ dbc,
                            const float* __restrict__ u2,
                            const float* __restrict__ xz,   // z = cols [512,1024)
                            const float* __restrict__ W_dt,
                            const float* __restrict__ b_dt,
                            const float* __restrict__ A_log,
                            const float* __restrict__ D_skip,
                            _Float16* __restrict__ yh) {
    __shared__ float sd[DT_RANK + 2 * D_STATE];    // 48
    const int b = blockIdx.x;
    const int d = threadIdx.x;                     // 0..511
    float wdt[DT_RANK], Av[D_STATE], h[D_STATE];
#pragma unroll
    for (int r = 0; r < DT_RANK; ++r) wdt[r] = W_dt[(size_t)r * D_INNER + d];
#pragma unroll
    for (int s = 0; s < D_STATE; ++s) { Av[s] = -expf(A_log[(size_t)d * D_STATE + s]); h[s] = 0.f; }
    const float bd = b_dt[d];
    const float Ds = D_skip[d];

    for (int l = 0; l < L_SEQ; ++l) {
        const int token = (b << 10) + l;
        if (d < DT_RANK + 2 * D_STATE)
            sd[d] = dbc[(size_t)token * (DT_RANK + 2 * D_STATE) + d];
        __syncthreads();
        float dtr = bd;
#pragma unroll
        for (int r = 0; r < DT_RANK; ++r) dtr += sd[r] * wdt[r];
        float dt = (dtr > 20.f) ? dtr : log1pf(expf(dtr));   // softplus
        float ut = u2[(size_t)token * D_INNER + d];
        float y = 0.f;
#pragma unroll
        for (int s = 0; s < D_STATE; ++s) {
            float dA = expf(dt * Av[s]);
            h[s] = dA * h[s] + dt * sd[DT_RANK + s] * ut;
            y += h[s] * sd[DT_RANK + D_STATE + s];
        }
        float z = xz[(size_t)token * (2 * D_INNER) + D_INNER + d];
        float g = z / (1.f + expf(-z));                      // SiLU(z)
        yh[(size_t)token * D_INNER + d] = (_Float16)((y + ut * Ds) * g);
        __syncthreads();
    }
}

// ---------------------------------------------------------------------------
// K6: out = yh(8192x512) @ W_out + x (residual), written to NCHW layout.
// ---------------------------------------------------------------------------
__global__ void gemm_out_kernel(const _Float16* __restrict__ A,
                                const _Float16* __restrict__ Bt,
                                const float* __restrict__ x,
                                float* __restrict__ out) {
    const int lane = threadIdx.x & 31;
    const int tile = blockIdx.x * 8 + (threadIdx.x >> 5);
    const int NT = DIM / 16;                       // 16
    const int mt = tile / NT, nt = tile % NT;
    v8f c = wmma_tile_f16(A, D_INNER, Bt, D_INNER, mt * 16, nt * 16, D_INNER, lane);
    const int cc = nt * 16 + (lane & 15);
    const int t0 = mt * 16 + ((lane >> 4) << 3);
#pragma unroll
    for (int r = 0; r < 8; ++r) {
        int token = t0 + r;
        int b = token >> 10, l = token & (L_SEQ - 1);
        size_t idx = ((size_t)b * DIM + cc) * L_SEQ + l;
        out[idx] = c[r] + x[idx];
    }
}

// ---------------------------------------------------------------------------
// launch
// ---------------------------------------------------------------------------
extern "C" void kernel_launch(void* const* d_in, const int* in_sizes, int n_in,
                              void* d_out, int out_size, void* d_ws, size_t ws_size,
                              hipStream_t stream) {
    const float* x       = (const float*)d_in[0];
    const float* ln_g    = (const float*)d_in[1];
    const float* ln_b    = (const float*)d_in[2];
    const float* W_in    = (const float*)d_in[3];
    const float* conv_w  = (const float*)d_in[4];
    const float* conv_b  = (const float*)d_in[5];
    const float* W_xproj = (const float*)d_in[6];
    const float* W_dt    = (const float*)d_in[7];
    const float* b_dt    = (const float*)d_in[8];
    const float* A_log   = (const float*)d_in[9];
    const float* D_skip  = (const float*)d_in[10];
    const float* W_out   = (const float*)d_in[11];
    float* out = (float*)d_out;

    // workspace carve-up (all 256B aligned)
    char* ws = (char*)d_ws;
    size_t off = 0;
    auto alloc = [&](size_t bytes) { void* p = ws + off; off = (off + bytes + 255) & ~(size_t)255; return p; };
    _Float16* WhinT  = (_Float16*)alloc((size_t)E_WIN  * 2);            // 512 KB
    _Float16* WhxpT  = (_Float16*)alloc((size_t)E_WXP  * 2);            //  48 KB
    _Float16* WhoutT = (_Float16*)alloc((size_t)E_WOUT * 2);            // 256 KB
    _Float16* xnh    = (_Float16*)alloc((size_t)N_TOK * DIM * 2);       //   4 MB
    float*    xz     = (float*)   alloc((size_t)N_TOK * 2*D_INNER * 4); //  32 MB
    float*    u2     = (float*)   alloc((size_t)N_TOK * D_INNER * 4);   //  16 MB
    _Float16* u2h    = (_Float16*)alloc((size_t)N_TOK * D_INNER * 2);   //   8 MB (reused as yh)
    float*    dbc    = (float*)   alloc((size_t)N_TOK * 48 * 4);        // 1.5 MB
    _Float16* yh     = u2h;  // u2h is dead after gemm_xproj; scan writes yh here

    // K0: weight convert+transpose
    convert_weights_kernel<<<(E_WIN + E_WXP + E_WOUT + 255) / 256, 256, 0, stream>>>(
        W_in, W_xproj, W_out, WhinT, WhxpT, WhoutT);
    // K1: layernorm -> f16
    layernorm_kernel<<<N_TOK, DIM, 0, stream>>>(x, ln_g, ln_b, xnh);
    // K2: xz = xn @ W_in   (512 x 64 tiles)
    gemm_in_kernel<<<(N_TOK / 16) * ((2 * D_INNER) / 16) / 8, 256, 0, stream>>>(xnh, WhinT, xz);
    // K3: depthwise conv + SiLU
    conv_silu_kernel<<<(BATCH * D_INNER) / 256, 256, 0, stream>>>(xz, conv_w, conv_b, u2, u2h);
    // K4: dbc = u2 @ W_xproj   (512 x 3 tiles)
    gemm_xproj_kernel<<<(N_TOK / 16) * 3 / 8, 256, 0, stream>>>(u2h, WhxpT, dbc);
    // K5: fused selective scan
    scan_kernel<<<BATCH, D_INNER, 0, stream>>>(dbc, u2, xz, W_dt, b_dt, A_log, D_skip, yh);
    // K6: out = y @ W_out + x   (512 x 16 tiles)
    gemm_out_kernel<<<(N_TOK / 16) * (DIM / 16) / 8, 256, 0, stream>>>(yh, WhoutT, x, out);
}
